// MultiHeadAttention_22978075034309
// MI455X (gfx1250) — compile-verified, matrix operand order checked by reading, not run on hardware
//
#include <hip/hip_runtime.h>
#include <hip/hip_bf16.h>

typedef __attribute__((ext_vector_type(16))) __bf16 v16bf;
typedef __attribute__((ext_vector_type(8)))  float  v8f;

#define BATCH  2
#define SEQ    2048
#define DMODEL 1024
#define NHEAD  16
#define DHEAD  64

struct alignas(16) BF8 { __bf16 b[8]; };

// ---------------------------------------------------------------------------
// WMMA helper: D = A(16x32 bf16) * B(32x16 bf16) + C(16x16 f32)
// ---------------------------------------------------------------------------
__device__ __forceinline__ v8f wmma_bf16(v16bf a, v16bf b, v8f c) {
  return __builtin_amdgcn_wmma_f32_16x16x32_bf16(
      /*neg_a=*/false, a, /*neg_b=*/false, b,
      /*c_mod=*/(short)0, c, /*reuse_a=*/false, /*reuse_b=*/false);
}

// 16x32 bf16 A/B fragment from a row-major bf16 matrix (tile origin = base).
// Wave32 layout (ISA 7.12.2): lane<16 -> K={0..7,16..23}, lane>=16 -> K={8..15,24..31}
__device__ __forceinline__ v16bf frag_ld_bf16(const __bf16* base, int ld) {
  const int lane = threadIdx.x & 31;
  const int m    = lane & 15;
  const int kb   = (lane >> 4) << 3;   // 0 or 8
  const __bf16* p = base + (size_t)m * ld + kb;
  BF8 t0 = *(const BF8*)(p);
  BF8 t1 = *(const BF8*)(p + 16);
  v16bf f;
#pragma unroll
  for (int i = 0; i < 8; ++i) { f[i] = t0.b[i]; f[i + 8] = t1.b[i]; }
  return f;
}

// Same fragment but source is f32 (global or LDS); convert inline to bf16.
__device__ __forceinline__ v16bf frag_ld_f32(const float* base, int ld) {
  const int lane = threadIdx.x & 31;
  const int m    = lane & 15;
  const int kb   = (lane >> 4) << 3;
  const float* p = base + (size_t)m * ld + kb;
  float4 a0 = *(const float4*)(p);
  float4 a1 = *(const float4*)(p + 4);
  float4 a2 = *(const float4*)(p + 16);
  float4 a3 = *(const float4*)(p + 20);
  v16bf f;
  f[0]  = (__bf16)a0.x; f[1]  = (__bf16)a0.y; f[2]  = (__bf16)a0.z; f[3]  = (__bf16)a0.w;
  f[4]  = (__bf16)a1.x; f[5]  = (__bf16)a1.y; f[6]  = (__bf16)a1.z; f[7]  = (__bf16)a1.w;
  f[8]  = (__bf16)a2.x; f[9]  = (__bf16)a2.y; f[10] = (__bf16)a2.z; f[11] = (__bf16)a2.w;
  f[12] = (__bf16)a3.x; f[13] = (__bf16)a3.y; f[14] = (__bf16)a3.z; f[15] = (__bf16)a3.w;
  return f;
}

// ---------------------------------------------------------------------------
// Kernel: transpose + convert weight W[K,N] f32 -> Wt[N,K] bf16
// ---------------------------------------------------------------------------
__global__ void wcvt_kernel(const float* __restrict__ W, __bf16* __restrict__ Wt,
                            int K, int N) {
  int idx = blockIdx.x * blockDim.x + threadIdx.x;  // over N*K, contiguous k for fixed n
  if (idx >= N * K) return;
  int n = idx / K;
  int k = idx - n * K;
  Wt[idx] = (__bf16)W[(size_t)k * N + n];
}

// ---------------------------------------------------------------------------
// Kernel: C = A(f32[M,K]) @ Wt(bf16[N,K])^T + bias, bf16 output.
// Each wave computes a 16x64 strip (4 accumulators) -> A fragment reused 4x.
// mode 0: C bf16 row-major [M,N]   (Q / K projections)
// mode 1: C scattered into Vt[b][h][dhead][s] bf16 (V projection, transposed)
// ---------------------------------------------------------------------------
__global__ __launch_bounds__(256) void gemm_a32_kernel(
    const float* __restrict__ A, const __bf16* __restrict__ Wt,
    const float* __restrict__ bias, __bf16* __restrict__ out,
    int M, int N, int K, int mode) {
  const int wave  = threadIdx.x >> 5;
  const int strip = blockIdx.x * 8 + wave;     // strips of 16 rows x 64 cols
  const int nsn   = N >> 6;                    // 64-col strips per row band
  const int trow  = (strip / nsn) << 4;
  const int tcol  = (strip % nsn) << 6;
  if (trow >= M) return;

  v8f acc[4];
#pragma unroll
  for (int t = 0; t < 4; ++t) acc[t] = v8f{0.f,0.f,0.f,0.f,0.f,0.f,0.f,0.f};

  for (int k0 = 0; k0 < K; k0 += 32) {
    v16bf a = frag_ld_f32(A + (size_t)trow * K + k0, K);
#pragma unroll
    for (int t = 0; t < 4; ++t) {
      v16bf b = frag_ld_bf16(Wt + (size_t)(tcol + 16 * t) * K + k0, K);
      acc[t] = wmma_bf16(a, b, acc[t]);
    }
  }

  const int lane = threadIdx.x & 31;
  const int n    = lane & 15;
  const int m0   = (lane >> 4) << 3;

#pragma unroll
  for (int t = 0; t < 4; ++t) {
    const int gc   = tcol + 16 * t + n;
    const float bv = bias[gc];
    if (mode == 0) {
#pragma unroll
      for (int r = 0; r < 8; ++r)
        out[(size_t)(trow + m0 + r) * N + gc] = (__bf16)(acc[t][r] + bv);
    } else {
      // rows index (b,s); cols index (h,dd); store Vt[((b*H+h)*DHEAD+dd)*SEQ + s]
      int gr = trow + m0;           // = b*SEQ + s  (8 consecutive s)
      int bb = gr / SEQ;
      int s  = gr - bb * SEQ;
      int hh = gc / DHEAD;
      int dd = gc - hh * DHEAD;
      BF8 pk;
#pragma unroll
      for (int r = 0; r < 8; ++r) pk.b[r] = (__bf16)(acc[t][r] + bv);
      __bf16* dst = out + ((size_t)(bb * NHEAD + hh) * DHEAD + dd) * SEQ + s;
      *(BF8*)dst = pk;
    }
  }
}

// ---------------------------------------------------------------------------
// Kernel: out(f32[M,N]) = A(bf16[M,K]) @ Wt(bf16[N,K])^T + bias   (out proj)
// Same 16x64 strip blocking.
// ---------------------------------------------------------------------------
__global__ __launch_bounds__(256) void gemm_a16_kernel(
    const __bf16* __restrict__ A, const __bf16* __restrict__ Wt,
    const float* __restrict__ bias, float* __restrict__ out,
    int M, int N, int K) {
  const int wave  = threadIdx.x >> 5;
  const int strip = blockIdx.x * 8 + wave;
  const int nsn   = N >> 6;
  const int trow  = (strip / nsn) << 4;
  const int tcol  = (strip % nsn) << 6;
  if (trow >= M) return;

  v8f acc[4];
#pragma unroll
  for (int t = 0; t < 4; ++t) acc[t] = v8f{0.f,0.f,0.f,0.f,0.f,0.f,0.f,0.f};

  for (int k0 = 0; k0 < K; k0 += 32) {
    v16bf a = frag_ld_bf16(A + (size_t)trow * K + k0, K);
#pragma unroll
    for (int t = 0; t < 4; ++t) {
      v16bf b = frag_ld_bf16(Wt + (size_t)(tcol + 16 * t) * K + k0, K);
      acc[t] = wmma_bf16(a, b, acc[t]);
    }
  }

  const int lane = threadIdx.x & 31;
  const int n    = lane & 15;
  const int m0   = (lane >> 4) << 3;
#pragma unroll
  for (int t = 0; t < 4; ++t) {
    const int gc   = tcol + 16 * t + n;
    const float bv = bias[gc];
#pragma unroll
    for (int r = 0; r < 8; ++r)
      out[(size_t)(trow + m0 + r) * N + gc] = acc[t][r] + bv;
  }
}

// ---------------------------------------------------------------------------
// Attention kernel: one (b, h, 16-row q-block) per workgroup (128 thr = 4 waves).
// LDS holds the full 16 x 2048 f32 score panel (128 KB of the WGP's 320 KB).
//  P1: scores = Q·K^T * 1/8 + mask*-1e9   (WMMA bf16, 64-col strips per wave)
//  P2: row softmax in f32 (v_exp_f32), attn -> d_out (float4 stores)
//  P3: ctx = attn @ V                      (WMMA, A from LDS, B from Vt)
// ---------------------------------------------------------------------------
__global__ __launch_bounds__(128) void attn_kernel(
    const __bf16* __restrict__ Qb, const __bf16* __restrict__ Kb,
    const __bf16* __restrict__ Vt, const float* __restrict__ mask,
    float* __restrict__ attn, __bf16* __restrict__ ctx) {
  extern __shared__ float smem[];
  float* sc   = smem;              // 16 * SEQ
  float* red  = sc + 16 * SEQ;     // 128
  float* rowm = red + 128;         // 16
  float* rows = rowm + 16;         // 16

  const int q0   = blockIdx.x << 4;
  const int h    = blockIdx.y;
  const int b    = blockIdx.z;
  const int wave = threadIdx.x >> 5;
  const int lane = threadIdx.x & 31;
  const int n    = lane & 15;
  const int m0   = (lane >> 4) << 3;

  // ---- Phase 1: score panel, 64-col strips ---------------------------------
  const __bf16* Qbase = Qb + ((size_t)(b * SEQ + q0)) * DMODEL + h * DHEAD;
  v16bf qa0 = frag_ld_bf16(Qbase + 0,  DMODEL);   // K-steps 0..31
  v16bf qa1 = frag_ld_bf16(Qbase + 32, DMODEL);   // K-steps 32..63

  for (int j = 0; j < (SEQ / 64) / 4; ++j) {      // 8 strips per wave
    const int col0 = (wave + 4 * j) << 6;
    v8f acc[4];
#pragma unroll
    for (int t = 0; t < 4; ++t) acc[t] = v8f{0.f,0.f,0.f,0.f,0.f,0.f,0.f,0.f};
#pragma unroll
    for (int t = 0; t < 4; ++t) {
      const __bf16* Kbase =
          Kb + ((size_t)(b * SEQ + col0 + 16 * t)) * DMODEL + h * DHEAD;
      acc[t] = wmma_bf16(qa0, frag_ld_bf16(Kbase + 0,  DMODEL), acc[t]);
      acc[t] = wmma_bf16(qa1, frag_ld_bf16(Kbase + 32, DMODEL), acc[t]);
    }
#pragma unroll
    for (int t = 0; t < 4; ++t) {
#pragma unroll
      for (int r = 0; r < 8; ++r) {
        const int row = m0 + r;
        const int col = col0 + 16 * t + n;
        float mval = mask[((size_t)b * SEQ + (q0 + row)) * SEQ + col];
        sc[row * SEQ + col] = acc[t][r] * 0.125f + mval * -1e9f;
      }
    }
  }
  __syncthreads();

  // ---- Phase 2: softmax (8 threads per row, 256 cols each) ------------------
  const int srow = threadIdx.x >> 3;
  const int sub  = threadIdx.x & 7;
  float* rp = sc + srow * SEQ + sub * 256;

  float lmax = -3.4e38f;
  for (int c = 0; c < 256; ++c) lmax = fmaxf(lmax, rp[c]);
  red[threadIdx.x] = lmax;
  __syncthreads();
  if (sub == 0) {
    float mv = red[srow * 8];
#pragma unroll
    for (int t = 1; t < 8; ++t) mv = fmaxf(mv, red[srow * 8 + t]);
    rowm[srow] = mv;
  }
  __syncthreads();
  const float mrow = rowm[srow];
  float lsum = 0.f;
  for (int c = 0; c < 256; ++c) {
    float e = __expf(rp[c] - mrow);
    rp[c] = e;
    lsum += e;
  }
  red[threadIdx.x] = lsum;
  __syncthreads();
  if (sub == 0) {
    float sv = 0.f;
#pragma unroll
    for (int t = 0; t < 8; ++t) sv += red[srow * 8 + t];
    rows[srow] = 1.f / sv;
  }
  __syncthreads();
  const float inv = rows[srow];
  float* gout = attn + (((size_t)b * NHEAD + h) * SEQ + (q0 + srow)) * SEQ + sub * 256;
  for (int c4 = 0; c4 < 64; ++c4) {
    float4 v = *(float4*)(rp + c4 * 4);
    v.x *= inv; v.y *= inv; v.z *= inv; v.w *= inv;
    *(float4*)(rp + c4 * 4)   = v;   // keep normalized attn in LDS for phase 3
    *(float4*)(gout + c4 * 4) = v;   // attn output
  }
  __syncthreads();

  // ---- Phase 3: ctx[16,64] = attn[16,2048] @ V[2048,64] ---------------------
  {
    const int cn = wave << 4;  // waves 0..3 cover DHEAD = 64
    const __bf16* Vbase = Vt + (((size_t)b * NHEAD + h) * DHEAD + cn) * SEQ;
    v8f acc = {0.f, 0.f, 0.f, 0.f, 0.f, 0.f, 0.f, 0.f};
    for (int ks = 0; ks < SEQ; ks += 32) {
      v16bf a  = frag_ld_f32(sc + ks, SEQ);        // A from LDS (ds_load), cvt bf16
      v16bf bb = frag_ld_bf16(Vbase + ks, SEQ);    // B contiguous (Vt layout)
      acc = wmma_bf16(a, bb, acc);
    }
#pragma unroll
    for (int r = 0; r < 8; ++r)
      ctx[((size_t)(b * SEQ) + q0 + m0 + r) * DMODEL + h * DHEAD + cn + n] =
          (__bf16)acc[r];
  }
}

// ---------------------------------------------------------------------------
// Host launcher
// ---------------------------------------------------------------------------
extern "C" void kernel_launch(void* const* d_in, const int* in_sizes, int n_in,
                              void* d_out, int out_size, void* d_ws, size_t ws_size,
                              hipStream_t stream) {
  (void)in_sizes; (void)n_in; (void)out_size; (void)ws_size;

  const float* v    = (const float*)d_in[0];
  const float* k    = (const float*)d_in[1];
  const float* q    = (const float*)d_in[2];
  const float* mask = (const float*)d_in[3];
  const float* wq   = (const float*)d_in[4];
  const float* bq   = (const float*)d_in[5];
  const float* wk   = (const float*)d_in[6];
  const float* bk   = (const float*)d_in[7];
  const float* wv   = (const float*)d_in[8];
  const float* bv   = (const float*)d_in[9];
  const float* wo   = (const float*)d_in[10];
  const float* bo   = (const float*)d_in[11];

  float* out  = (float*)d_out;                               // [B,S,D]
  float* attn = out + (size_t)BATCH * SEQ * DMODEL;          // [B,H,S,S]

  const size_t WBYTES = (size_t)DMODEL * DMODEL * sizeof(__bf16);      // 2 MB
  const size_t ABYTES = (size_t)BATCH * SEQ * DMODEL * sizeof(__bf16); // 8 MB
  char* ws = (char*)d_ws;
  __bf16* Wqt = (__bf16*)(ws);                 ws += WBYTES;
  __bf16* Wkt = (__bf16*)(ws);                 ws += WBYTES;
  __bf16* Wvt = (__bf16*)(ws);                 ws += WBYTES;
  __bf16* Wot = (__bf16*)(ws);                 ws += WBYTES;
  __bf16* Qb  = (__bf16*)(ws);                 ws += ABYTES;
  __bf16* Kbf = (__bf16*)(ws);                 ws += ABYTES;
  __bf16* Vtp = (__bf16*)(ws);                 ws += ABYTES;  // [B,H,DHEAD,SEQ]
  __bf16* ctx = (__bf16*)(ws);                 ws += ABYTES;  // [B,S,D]

  const int M = BATCH * SEQ;        // 4096
  const int N = DMODEL;             // 1024
  const int K = DMODEL;             // 1024

  // 1) weight transpose + bf16 convert
  {
    int ne = DMODEL * DMODEL;
    int blk = (ne + 255) / 256;
    wcvt_kernel<<<blk, 256, 0, stream>>>(wq, Wqt, DMODEL, DMODEL);
    wcvt_kernel<<<blk, 256, 0, stream>>>(wk, Wkt, DMODEL, DMODEL);
    wcvt_kernel<<<blk, 256, 0, stream>>>(wv, Wvt, DMODEL, DMODEL);
    wcvt_kernel<<<blk, 256, 0, stream>>>(wo, Wot, DMODEL, DMODEL);
  }

  // 2) projections: 16x64 strips, 8 strips/block -> 512 blocks
  const int gblocks = (M / 16) * (N / 64) / 8;  // 512
  gemm_a32_kernel<<<gblocks, 256, 0, stream>>>(q, Wqt, bq, Qb,  M, N, K, 0);
  gemm_a32_kernel<<<gblocks, 256, 0, stream>>>(k, Wkt, bk, Kbf, M, N, K, 0);
  gemm_a32_kernel<<<gblocks, 256, 0, stream>>>(v, Wvt, bv, Vtp, M, N, K, 1);

  // 3) attention (scores + softmax + attn output + context)
  {
    size_t dyn = (size_t)(16 * SEQ + 128 + 16 + 16) * sizeof(float);  // 131712 B
    (void)hipFuncSetAttribute((const void*)attn_kernel,
                              hipFuncAttributeMaxDynamicSharedMemorySize,
                              (int)dyn);
    dim3 grid(SEQ / 16, NHEAD, BATCH);
    attn_kernel<<<grid, 128, dyn, stream>>>(Qb, Kbf, Vtp, mask, attn, ctx);
  }

  // 4) output projection
  gemm_a16_kernel<<<gblocks, 256, 0, stream>>>(ctx, Wot, bo, out, M, N, K);
}
